// ProjectedGINRegressor_87265145520190
// MI455X (gfx1250) — matrix operand order repaired and checked by reading, not compiled, not run on hardware
//
#include <hip/hip_runtime.h>
#include <hip/hip_bf16.h>

#define GNN_N 50000
#define GNN_INC 128
#define GNN_HID 256

typedef __attribute__((ext_vector_type(16))) __bf16 v16bf;
typedef __attribute__((ext_vector_type(8)))  __bf16 v8bf;
typedef __attribute__((ext_vector_type(8)))  float  v8f;

// ---------------------------------------------------------------------------
// Pack f32 weights W[C x H] into per-lane WMMA B-fragment layout (bf16):
//   out[((tn*ktiles + kt)*32 + lane)*16 + i] = W[(kt*32 + (i>>3)*16 + hi*8 + (i&7))*H + tn*16 + r]
// hi = lane>>4, r = lane&15. Each lane's fragment is one contiguous 32B load.
// ---------------------------------------------------------------------------
__global__ __launch_bounds__(256) void pack_w_bf16(const float* __restrict__ W,
                                                   __bf16* __restrict__ out,
                                                   int C, int H) {
    int idx = blockIdx.x * blockDim.x + threadIdx.x;
    int total = C * H;
    if (idx >= total) return;
    int ktiles = C >> 5;
    int tn   = idx / (ktiles * 512);
    int rem  = idx % (ktiles * 512);
    int kt   = rem / 512;
    int q    = rem % 512;
    int lane = q >> 4;
    int i    = q & 15;
    int hi   = lane >> 4;
    int r    = lane & 15;
    int K    = kt * 32 + (i >> 3) * 16 + hi * 8 + (i & 7);
    int col  = tn * 16 + r;
    out[idx] = (__bf16)W[K * H + col];
}

// ---------------------------------------------------------------------------
// Edge scatter-add: z[dst] += h[src]   (z pre-initialized to h => z = h + agg)
// C4 = C/4 is a compile-time constant: channel mapping is shift/mask only,
// loop strides over edges. Threads tid 0..C4-1 cover one edge's full row,
// so the float4 gather is a fully-coalesced contiguous row read.
// ---------------------------------------------------------------------------
template <int C4>
__global__ __launch_bounds__(256) void scatter_add(float* __restrict__ z,
                                                   const float* __restrict__ h,
                                                   const int* __restrict__ src,
                                                   const int* __restrict__ dst,
                                                   int nE) {
    constexpr int EPB = 256 / C4;                 // edge slots per block
    const int c    = (threadIdx.x % C4) * 4;      // channel offset (compile-time mask)
    const int slot = threadIdx.x / C4;            // edge slot in block
    const long long estride = (long long)gridDim.x * EPB;
    for (long long e = (long long)blockIdx.x * EPB + slot; e < nE; e += estride) {
        // stream-prefetch next sweep's index words (speculative, OOB-safe)
        __builtin_prefetch(src + e + estride, 0, 3);
        __builtin_prefetch(dst + e + estride, 0, 3);
        int s = src[e];
        int d = dst[e];
        float4 v = *(const float4*)(h + (size_t)s * (C4 * 4) + c);
        float* zp = z + (size_t)d * (C4 * 4) + c;
        __hip_atomic_fetch_add(zp + 0, v.x, __ATOMIC_RELAXED, __HIP_MEMORY_SCOPE_AGENT);
        __hip_atomic_fetch_add(zp + 1, v.y, __ATOMIC_RELAXED, __HIP_MEMORY_SCOPE_AGENT);
        __hip_atomic_fetch_add(zp + 2, v.z, __ATOMIC_RELAXED, __HIP_MEMORY_SCOPE_AGENT);
        __hip_atomic_fetch_add(zp + 3, v.w, __ATOMIC_RELAXED, __HIP_MEMORY_SCOPE_AGENT);
    }
}

// ---------------------------------------------------------------------------
// A-fragment loaders (ISA 7.12.2: 16-bit A 16x32; lane hi-half holds K+8 / K+24)
// ---------------------------------------------------------------------------
__device__ inline v16bf load_a_frag(const float* __restrict__ A, int rowBase, int k0, int hi) {
    const float* p = A + rowBase + k0 + hi * 8;
    float4 f0 = ((const float4*)p)[0];
    float4 f1 = ((const float4*)p)[1];
    float4 f2 = ((const float4*)(p + 16))[0];
    float4 f3 = ((const float4*)(p + 16))[1];
    v16bf a;
    a[0]  = (__bf16)f0.x; a[1]  = (__bf16)f0.y; a[2]  = (__bf16)f0.z; a[3]  = (__bf16)f0.w;
    a[4]  = (__bf16)f1.x; a[5]  = (__bf16)f1.y; a[6]  = (__bf16)f1.z; a[7]  = (__bf16)f1.w;
    a[8]  = (__bf16)f2.x; a[9]  = (__bf16)f2.y; a[10] = (__bf16)f2.z; a[11] = (__bf16)f2.w;
    a[12] = (__bf16)f3.x; a[13] = (__bf16)f3.y; a[14] = (__bf16)f3.z; a[15] = (__bf16)f3.w;
    return a;
}

__device__ inline v16bf load_a_frag(const __bf16* __restrict__ A, int rowBase, int k0, int hi) {
    v8bf lo = *(const v8bf*)(A + rowBase + k0 + hi * 8);
    v8bf hh = *(const v8bf*)(A + rowBase + k0 + 16 + hi * 8);
    v16bf a;
#pragma unroll
    for (int i = 0; i < 8; ++i) { a[i] = lo[i]; a[8 + i] = hh[i]; }
    return a;
}

// ---------------------------------------------------------------------------
// WMMA GEMM:  out[MxH] = act(A[MxC] @ W[CxH] + bias), W pre-packed bf16 frags.
// One wave computes a 16x64 strip. Per k-step: 1 A frag, then ALL 4 B frags
// issued as one load clause, then 4 back-to-back WMMAs (independent
// accumulators -> no D->A/B hazards, loads overlap matrix pipe).
// ---------------------------------------------------------------------------
template <typename AT, typename OT, bool RELU>
__global__ __launch_bounds__(256) void gemm_wmma_bf16(const AT* __restrict__ A,
                                                      const __bf16* __restrict__ Wp,
                                                      const float* __restrict__ bias,
                                                      OT* __restrict__ out,
                                                      int nRows, int C, int H) {
    const int wave     = blockIdx.x * (blockDim.x >> 5) + (threadIdx.x >> 5);
    const int lane     = threadIdx.x & 31;
    const int tilesN64 = H >> 6;
    const int tilesM   = nRows >> 4;
    if (wave >= tilesM * tilesN64) return;   // wave-uniform: EXEC stays all-ones
    const int tm   = wave / tilesN64;
    const int tn64 = wave % tilesN64;

    const int r       = lane & 15;
    const int hi      = lane >> 4;
    const int rowBase = (tm * 16 + r) * C;
    const int ktiles  = C >> 5;

    v8f acc[4] = {};
    const __bf16* wlane = Wp + (size_t)(tn64 * 4) * ktiles * 512 + lane * 16;

    for (int kt = 0; kt < ktiles; ++kt) {
        v16bf a = load_a_frag(A, rowBase, kt * 32, hi);
        v16bf b0 = *(const v16bf*)(wlane + (size_t)(0 * ktiles + kt) * 512);
        v16bf b1 = *(const v16bf*)(wlane + (size_t)(1 * ktiles + kt) * 512);
        v16bf b2 = *(const v16bf*)(wlane + (size_t)(2 * ktiles + kt) * 512);
        v16bf b3 = *(const v16bf*)(wlane + (size_t)(3 * ktiles + kt) * 512);
        acc[0] = __builtin_amdgcn_wmma_f32_16x16x32_bf16(false, a, false, b0,
                                                         (short)0, acc[0], false, false);
        acc[1] = __builtin_amdgcn_wmma_f32_16x16x32_bf16(false, a, false, b1,
                                                         (short)0, acc[1], false, false);
        acc[2] = __builtin_amdgcn_wmma_f32_16x16x32_bf16(false, a, false, b2,
                                                         (short)0, acc[2], false, false);
        acc[3] = __builtin_amdgcn_wmma_f32_16x16x32_bf16(false, a, false, b3,
                                                         (short)0, acc[3], false, false);
    }

#pragma unroll
    for (int t = 0; t < 4; ++t) {
        const int col  = tn64 * 64 + t * 16 + r;
        const float bc = bias[col];
#pragma unroll
        for (int j = 0; j < 8; ++j) {
            int row = tm * 16 + j + hi * 8;   // C/D layout: lane-hi half = M+8
            float v = acc[t][j] + bc;
            if (RELU) v = v > 0.0f ? v : 0.0f;
            out[row * H + col] = (OT)v;
        }
    }
}

// ---------------------------------------------------------------------------
// Head: out[n] = h[n,:] . Wout + bout   (one wave32 per row)
// ---------------------------------------------------------------------------
__global__ __launch_bounds__(256) void head_dot(const float* __restrict__ h,
                                                const float* __restrict__ Wout,
                                                const float* __restrict__ bout,
                                                float* __restrict__ out,
                                                int nRows, int C) {
    int wave = blockIdx.x * (blockDim.x >> 5) + (threadIdx.x >> 5);
    int lane = threadIdx.x & 31;
    if (wave >= nRows) return;
    float s = 0.0f;
    const float* row = h + (long long)wave * C;
    for (int c = lane; c < C; c += 32) s += row[c] * Wout[c];
#pragma unroll
    for (int off = 16; off > 0; off >>= 1) s += __shfl_xor(s, off, 32);
    if (lane == 0) out[wave] = s + bout[0];
}

// ---------------------------------------------------------------------------
// Launch
// ---------------------------------------------------------------------------
extern "C" void kernel_launch(void* const* d_in, const int* in_sizes, int n_in,
                              void* d_out, int out_size, void* d_ws, size_t ws_size,
                              hipStream_t stream) {
    const float* x  = (const float*)d_in[0];
    const int*   ei = (const int*)d_in[1];
    const int    nE = in_sizes[1] / 2;

    // Workspace layout
    char* ws = (char*)d_ws;
    float*  bufA = (float*)ws;                                  // h / h_next (N x 256 f32)
    float*  bufB = (float*)(ws + (size_t)GNN_N * GNN_HID * 4);  // z          (N x 256 f32)
    __bf16* bufT = (__bf16*)(ws + (size_t)GNN_N * GNN_HID * 8); // t1         (N x 256 bf16)
    size_t off = (size_t)GNN_N * GNN_HID * 10;

    // packed bf16 weights: order W1_0, W2_0, W1_1, W2_1, W1_2, W2_2
    const int widx[6]  = {2, 4, 6, 8, 10, 12};
    const int wC[6]    = {GNN_INC, GNN_HID, GNN_HID, GNN_HID, GNN_HID, GNN_HID};
    __bf16* wb[6];
    for (int j = 0; j < 6; ++j) {
        int total = wC[j] * GNN_HID;
        wb[j] = (__bf16*)(ws + off);
        off += (size_t)total * 2;
        off = (off + 31) & ~(size_t)31;
        pack_w_bf16<<<(total + 255) / 256, 256, 0, stream>>>(
            (const float*)d_in[widx[j]], wb[j], wC[j], GNN_HID);
    }

    const int waves = (GNN_N / 16) * (GNN_HID / 64);  // 12500 waves (16x64 strips)
    const int gblks = (waves + 7) / 8;                // 8 waves / block

    const float* h = x;
    for (int l = 0; l < 3; ++l) {
        const int C = (l == 0) ? GNN_INC : GNN_HID;
        // z = h
        hipMemcpyAsync(bufB, h, (size_t)GNN_N * C * sizeof(float),
                       hipMemcpyDeviceToDevice, stream);
        // z += sum_{edges} h[src] -> dst
        if (C == GNN_INC)
            scatter_add<GNN_INC / 4><<<4096, 256, 0, stream>>>(bufB, h, ei, ei + nE, nE);
        else
            scatter_add<GNN_HID / 4><<<4096, 256, 0, stream>>>(bufB, h, ei, ei + nE, nE);
        // t1 = relu(z @ W1 + b1)   (bf16 out)
        gemm_wmma_bf16<float, __bf16, true><<<gblks, 256, 0, stream>>>(
            bufB, wb[2 * l], (const float*)d_in[3 + 4 * l], bufT, GNN_N, C, GNN_HID);
        // h = relu(t1 @ W2 + b2)   (outer relu, f32 out)
        gemm_wmma_bf16<__bf16, float, true><<<gblks, 256, 0, stream>>>(
            bufT, wb[2 * l + 1], (const float*)d_in[5 + 4 * l], bufA, GNN_N, GNN_HID, GNN_HID);
        h = bufA;
    }

    // head: out = h @ W_out + b_out
    head_dot<<<(GNN_N + 7) / 8, 256, 0, stream>>>(
        bufA, (const float*)d_in[14], (const float*)d_in[15], (float*)d_out,
        GNN_N, GNN_HID);
}
